// MoonshineStreamingEncoder_1133871366233
// MI455X (gfx1250) — matrix operand order repaired and checked by reading, not compiled, not run on hardware
//
#include <hip/hip_runtime.h>
#include <hip/hip_bf16.h>
#include <math.h>

#define B_    4
#define AUDIO_ 327680
#define FRAME_ 80
#define TF_   4096
#define T1_   2048
#define TENC_ 1024
#define HID_  320
#define C1_   640
#define NH_   8
#define NKV_  4
#define HD_   40
#define FFN_  1280
#define NL_   6

typedef __attribute__((ext_vector_type(16))) __bf16 v16bf;
typedef __attribute__((ext_vector_type(8)))  __bf16 v8bf;
typedef __attribute__((ext_vector_type(8)))  float  v8f;

__device__ __forceinline__ __bf16 f2bf(float f) {
  unsigned u = __builtin_bit_cast(unsigned, f);
  unsigned r = u + 0x7fffu + ((u >> 16) & 1u);   // round-to-nearest-even
  unsigned short hs = (unsigned short)(r >> 16);
  return __builtin_bit_cast(__bf16, hs);
}
__device__ __forceinline__ __bf16 bf_zero() {
  return __builtin_bit_cast(__bf16, (unsigned short)0);
}
__device__ __forceinline__ float silu_f(float x) { return x / (1.f + __expf(-x)); }

// ---------------------------------------------------------------------------
// fp32 -> bf16 bulk convert (weights)
// ---------------------------------------------------------------------------
__global__ __launch_bounds__(256) void f32_to_bf16_kernel(const float* __restrict__ s,
                                                          __bf16* __restrict__ d, int n) {
  int i = blockIdx.x * 256 + threadIdx.x;
  if (i < n) d[i] = f2bf(s[i]);
}

// ---------------------------------------------------------------------------
// num_frames + prefix-mask lengths through the two strided causal convs
// nf[0..3] = frames, nf[4..7] = valid conv1 rows, nf[8..11] = valid conv2 rows
// ---------------------------------------------------------------------------
__global__ __launch_bounds__(256) void nf_kernel(const unsigned char* __restrict__ pm,
                                                 int* __restrict__ nf) {
  __shared__ int red[256];
  int b = blockIdx.x, tid = threadIdx.x;
  int s = 0;
  for (int i = tid; i < AUDIO_; i += 256) s += (pm[(size_t)b * AUDIO_ + i] != 0);
  red[tid] = s; __syncthreads();
  for (int off = 128; off > 0; off >>= 1) {
    if (tid < off) red[tid] += red[tid + off];
    __syncthreads();
  }
  if (tid == 0) {
    int frames = red[0] / FRAME_;
    int n1 = frames > 0 ? ((frames + 3) / 2 + 1) : 0; if (n1 > T1_)   n1 = T1_;
    int n2 = n1 > 0     ? ((n1 + 3) / 2 + 1)     : 0; if (n2 > TENC_) n2 = TENC_;
    nf[b] = frames; nf[4 + b] = n1; nf[8 + b] = n2;
  }
}

// ---------------------------------------------------------------------------
// frontend: per-frame CMVN -> asinh(exp(log_k)*x) -> linear 80->320 -> silu -> mask
// writes bf16 (consumed by conv1 implicit GEMM)
// ---------------------------------------------------------------------------
__global__ __launch_bounds__(320) void frontend_kernel(
    const float* __restrict__ audio, const float* __restrict__ lin_w,
    const float* __restrict__ log_k, const int* __restrict__ nf0,
    __bf16* __restrict__ x1) {
  __shared__ float frm[FRAME_];
  __shared__ float smu, srs;
  int fb = blockIdx.x;
  int b = fb / TF_, t = fb - b * TF_;
  int tid = threadIdx.x;
  if (tid < FRAME_) frm[tid] = audio[(size_t)b * AUDIO_ + (size_t)t * FRAME_ + tid];
  __syncthreads();
  if (tid == 0) {
    float s = 0.f;
    for (int i = 0; i < FRAME_; ++i) s += frm[i];
    float mu = s / (float)FRAME_;
    float vv = 0.f;
    for (int i = 0; i < FRAME_; ++i) { float c = frm[i] - mu; vv += c * c; }
    vv /= (float)FRAME_;
    smu = mu; srs = rsqrtf(vv + 1e-6f);
  }
  __syncthreads();
  float kk = __expf(log_k[0]);
  if (tid < FRAME_) frm[tid] = asinhf(kk * (frm[tid] - smu) * srs);
  __syncthreads();
  float acc = 0.f;
  for (int i = 0; i < FRAME_; ++i) acc += frm[i] * lin_w[tid * FRAME_ + i];
  acc = silu_f(acc);
  if (t >= nf0[b]) acc = 0.f;
  x1[(size_t)fb * HID_ + tid] = f2bf(acc);
}

// ---------------------------------------------------------------------------
// LayerNorm (no bias), one block per row, 320 threads; fp32 in, bf16 out
// ---------------------------------------------------------------------------
__global__ __launch_bounds__(320) void ln_kernel_bf(const float* __restrict__ x,
                                                    const float* __restrict__ w,
                                                    __bf16* __restrict__ y, float eps) {
  __shared__ float buf[HID_];
  __shared__ float smu, srs;
  int m = blockIdx.x, c = threadIdx.x;
  float val = x[(size_t)m * HID_ + c];
  buf[c] = val;
  __syncthreads();
  if (c == 0) {
    float s = 0.f;
    for (int i = 0; i < HID_; ++i) s += buf[i];
    float mu = s / (float)HID_;
    float vv = 0.f;
    for (int i = 0; i < HID_; ++i) { float d = buf[i] - mu; vv += d * d; }
    vv /= (float)HID_;
    smu = mu; srs = rsqrtf(vv + eps);
  }
  __syncthreads();
  y[(size_t)m * HID_ + c] = f2bf((val - smu) * srs * w[c]);
}

// fp32 in / fp32 out variant (final LN)
__global__ __launch_bounds__(320) void ln_kernel_f32(const float* __restrict__ x,
                                                     const float* __restrict__ w,
                                                     float* __restrict__ y, float eps) {
  __shared__ float buf[HID_];
  __shared__ float smu, srs;
  int m = blockIdx.x, c = threadIdx.x;
  float val = x[(size_t)m * HID_ + c];
  buf[c] = val;
  __syncthreads();
  if (c == 0) {
    float s = 0.f;
    for (int i = 0; i < HID_; ++i) s += buf[i];
    float mu = s / (float)HID_;
    float vv = 0.f;
    for (int i = 0; i < HID_; ++i) { float d = buf[i] - mu; vv += d * d; }
    vv /= (float)HID_;
    smu = mu; srs = rsqrtf(vv + eps);
  }
  __syncthreads();
  y[(size_t)m * HID_ + c] = (val - smu) * srs * w[c];
}

// ---------------------------------------------------------------------------
// WMMA implicit-GEMM / causal-conv kernel, bf16 x bf16 -> fp32 accumulate.
//   patch(b,t,kg): j = kg/Cin, ci = kg%Cin, tin = t*strd + j - lpad (0 if OOB)
//   plain GEMM: K5=1, strd=1, lpad=0 -> tin = t, Wb is [Cout, Cin] row-major.
// Block 256 threads = 8 waves; tile 128x64; each wave owns 32x32 (4 WMMA/step).
// In-bounds tiles are staged LDS-direct via global_load_async_to_lds_b128
// (ASYNCcnt); conv patches / ragged edges use element-gather fallback.
// Epilogue: (+bias) -> zero rows t >= nfrow[b] -> silu -> (+res fp32)
//           -> store fp32 (Yf) and/or bf16 (Yb).
// ---------------------------------------------------------------------------
__global__ __launch_bounds__(256) void gemm_conv_wmma(
    const __bf16* __restrict__ X, const __bf16* __restrict__ Wb,
    const float* __restrict__ bias, const float* __restrict__ res,
    float* __restrict__ Yf, __bf16* __restrict__ Yb,
    const int* __restrict__ nfrow,
    int Bn, int Tin, int Tout, int Cin, int Cout,
    int K5, int strd, int lpad, int Ktot, int doSilu) {
  __shared__ __bf16 As[128][40];   // pitch 80B -> 16B-aligned subrows
  __shared__ __bf16 Bst[64][40];   // B stored transposed: [n][k]
  const int M = Bn * Tout;
  const int m0 = blockIdx.y * 128;
  const int n0 = blockIdx.x * 64;
  const int tid = threadIdx.x;
  const int lane = tid & 31;
  const int wave = tid >> 5;
  const int wr = (wave >> 1) << 5;   // 0,32,64,96
  const int wc = (wave & 1) << 5;    // 0,32
  const bool gemmFast = (K5 == 1) && (strd == 1) && (lpad == 0) &&
                        ((Ktot & 31) == 0) && (m0 + 128 <= M);
  const bool bFast = gemmFast && (n0 + 64 <= Cout);

  v8f acc00 = {}, acc01 = {}, acc10 = {}, acc11 = {};

  for (int k0 = 0; k0 < Ktot; k0 += 32) {
    // ---- stage A tile (128 x 32) ----
    if (gemmFast) {
#pragma unroll
      for (int i = 0; i < 2; ++i) {
        int c = tid + i * 256;            // 512 chunks of 8 bf16
        int r = c >> 2, ck = (c & 3) << 3;
        unsigned lds = (unsigned)(unsigned long long)(&As[r][ck]);
        unsigned long long ga =
            (unsigned long long)(X + (size_t)(m0 + r) * Cin + (k0 + ck));
        asm volatile("global_load_async_to_lds_b128 %0, %1, off"
                     :: "v"(lds), "v"(ga) : "memory");
      }
    } else {
#pragma unroll
      for (int i = 0; i < 16; ++i) {
        int idx = tid + i * 256;
        int r = idx >> 5, kk = idx & 31;
        int m = m0 + r, kg = k0 + kk;
        __bf16 val = bf_zero();
        if (m < M && kg < Ktot) {
          int b = m / Tout, t = m - b * Tout;
          int j = kg / Cin, ci = kg - j * Cin;
          int tin = t * strd + j - lpad;
          if (tin >= 0 && tin < Tin)
            val = X[((size_t)(b * Tin + tin)) * (size_t)Cin + ci];
        }
        As[r][kk] = val;
      }
    }
    // ---- stage B tile transposed (64 n x 32 k) ----
    if (bFast) {
      int n = tid >> 2, ck = (tid & 3) << 3;    // 256 chunks of 8 bf16
      unsigned lds = (unsigned)(unsigned long long)(&Bst[n][ck]);
      unsigned long long ga =
          (unsigned long long)(Wb + (size_t)(n0 + n) * Cin + (k0 + ck));
      asm volatile("global_load_async_to_lds_b128 %0, %1, off"
                   :: "v"(lds), "v"(ga) : "memory");
    } else {
#pragma unroll
      for (int i = 0; i < 8; ++i) {
        int idx = tid + i * 256;
        int n = idx >> 5, kk = idx & 31;
        int kg = k0 + kk, ng = n0 + n;
        __bf16 val = bf_zero();
        if (ng < Cout && kg < Ktot) {
          int j = kg / Cin, ci = kg - j * Cin;
          val = Wb[((size_t)ng * Cin + ci) * (size_t)K5 + j];
        }
        Bst[n][kk] = val;
      }
    }
    if (gemmFast || bFast)
      asm volatile("s_wait_asynccnt 0x0" ::: "memory");
    __syncthreads();

    // ---- fragments (128-bit LDS loads, ISA VGPR layouts) ----
    const int ml0 = wr + (lane & 15);
    const int ml1 = ml0 + 16;
    const int kb = (lane >> 4) << 3;    // A: K base 0 / 8
    v8bf alo = *(const v8bf*)&As[ml0][kb];
    v8bf ahi = *(const v8bf*)&As[ml0][kb + 16];
    v16bf a0 = __builtin_shufflevector(alo, ahi, 0, 1, 2, 3, 4, 5, 6, 7,
                                       8, 9, 10, 11, 12, 13, 14, 15);
    alo = *(const v8bf*)&As[ml1][kb];
    ahi = *(const v8bf*)&As[ml1][kb + 16];
    v16bf a1 = __builtin_shufflevector(alo, ahi, 0, 1, 2, 3, 4, 5, 6, 7,
                                       8, 9, 10, 11, 12, 13, 14, 15);
    const int nl = wc + (lane & 15);
    const int kb2 = (lane >> 4) << 3;   // B: K base 0 / 8
    v8bf blo = *(const v8bf*)&Bst[nl][kb2];
    v8bf bhi = *(const v8bf*)&Bst[nl][kb2 + 16];
    v16bf b0 = __builtin_shufflevector(blo, bhi, 0, 1, 2, 3, 4, 5, 6, 7,
                                       8, 9, 10, 11, 12, 13, 14, 15);
    blo = *(const v8bf*)&Bst[nl + 16][kb2];
    bhi = *(const v8bf*)&Bst[nl + 16][kb2 + 16];
    v16bf b1 = __builtin_shufflevector(blo, bhi, 0, 1, 2, 3, 4, 5, 6, 7,
                                       8, 9, 10, 11, 12, 13, 14, 15);

    acc00 = __builtin_amdgcn_wmma_f32_16x16x32_bf16(false, a0, false, b0, (short)0, acc00, false, false);
    acc01 = __builtin_amdgcn_wmma_f32_16x16x32_bf16(false, a0, false, b1, (short)0, acc01, false, false);
    acc10 = __builtin_amdgcn_wmma_f32_16x16x32_bf16(false, a1, false, b0, (short)0, acc10, false, false);
    acc11 = __builtin_amdgcn_wmma_f32_16x16x32_bf16(false, a1, false, b1, (short)0, acc11, false, false);
    __syncthreads();
  }

  // ---- epilogue: C layout VGPR v -> M = v + 8*(lane>=16), N = lane&15 ----
  v8f accs[2][2] = {{acc00, acc01}, {acc10, acc11}};
#pragma unroll
  for (int rt = 0; rt < 2; ++rt) {
#pragma unroll
    for (int v = 0; v < 8; ++v) {
      int m = m0 + wr + (rt << 4) + v + ((lane >> 4) << 3);
      if (m >= M) continue;
      int b = m / Tout, t = m - b * Tout;
      bool rowok = (nfrow == nullptr) || (t < nfrow[b]);
#pragma unroll
      for (int ct = 0; ct < 2; ++ct) {
        int n = n0 + wc + (ct << 4) + (lane & 15);
        if (n >= Cout) continue;
        float val = accs[rt][ct][v];
        if (bias) val += bias[n];
        if (!rowok) val = 0.f;
        if (doSilu) val = silu_f(val);
        if (res) val += res[(size_t)m * Cout + n];
        if (Yf) Yf[(size_t)m * Cout + n] = val;
        if (Yb) Yb[(size_t)m * Cout + n] = f2bf(val);
      }
    }
  }
}

// ---------------------------------------------------------------------------
// Windowed GQA attention, one thread per (b, head, query).
// q/k/v fp32: q [B*T, NH*HD], k/v [B*T, NKV*HD]; out bf16 [B*T, NH*HD].
// ---------------------------------------------------------------------------
__global__ __launch_bounds__(256) void attn_kernel(
    const float* __restrict__ q, const float* __restrict__ k,
    const float* __restrict__ v, __bf16* __restrict__ o,
    const int* __restrict__ nf2, int lw, int rw) {
  int gid = blockIdx.x * blockDim.x + threadIdx.x;
  if (gid >= B_ * TENC_ * NH_) return;
  int h = gid % NH_;
  int tq = (gid / NH_) % TENC_;
  int b = gid / (NH_ * TENC_);
  int hk = h >> 1;                           // n_rep = NH/NKV = 2
  const float scale = 0.15811388300841897f;  // 1/sqrt(40)

  float qv[HD_];
  const float* qp = q + ((size_t)(b * TENC_ + tq) * NH_ + h) * HD_;
#pragma unroll
  for (int d = 0; d < HD_; ++d) qv[d] = qp[d];

  __bf16* op = o + ((size_t)(b * TENC_ + tq) * NH_ + h) * HD_;

  int kmax = nf2[b]; if (kmax > TENC_) kmax = TENC_;
  int klo = tq - (lw - 1); if (klo < 0) klo = 0;
  int khi = tq + (rw > 0 ? rw - 1 : 0); if (khi > kmax - 1) khi = kmax - 1;
  if (khi < klo) {
#pragma unroll
    for (int d = 0; d < HD_; ++d) op[d] = bf_zero();
    return;
  }
  int cnt = khi - klo + 1;
  if (cnt > 24) cnt = 24;

  float sc[24];
  float mx = -3.4e38f;
  for (int i = 0; i < cnt; ++i) {
    const float* kp = k + ((size_t)(b * TENC_ + klo + i) * NKV_ + hk) * HD_;
    float s = 0.f;
#pragma unroll
    for (int d = 0; d < HD_; ++d) s += qv[d] * kp[d];
    s *= scale;
    sc[i] = s;
    if (s > mx) mx = s;
  }
  float den = 0.f;
  for (int i = 0; i < cnt; ++i) { float e = __expf(sc[i] - mx); sc[i] = e; den += e; }
  float inv = 1.f / den;

  float acc[HD_];
#pragma unroll
  for (int d = 0; d < HD_; ++d) acc[d] = 0.f;
  for (int i = 0; i < cnt; ++i) {
    const float* vp = v + ((size_t)(b * TENC_ + klo + i) * NKV_ + hk) * HD_;
    float p = sc[i] * inv;
#pragma unroll
    for (int d = 0; d < HD_; ++d) acc[d] += p * vp[d];
  }
#pragma unroll
  for (int d = 0; d < HD_; ++d) op[d] = f2bf(acc[d]);
}

// ---------------------------------------------------------------------------
extern "C" void kernel_launch(void* const* d_in, const int* in_sizes, int n_in,
                              void* d_out, int out_size, void* d_ws, size_t ws_size,
                              hipStream_t stream) {
  (void)in_sizes; (void)n_in; (void)out_size; (void)ws_size;
  const float* audio   = (const float*)d_in[0];
  const unsigned char* pmask = (const unsigned char*)d_in[1];
  const float* log_k   = (const float*)d_in[2];
  const float* lin_w   = (const float*)d_in[3];
  const float* conv1_w = (const float*)d_in[4];
  const float* conv1_b = (const float*)d_in[5];
  const float* conv2_w = (const float*)d_in[6];
  const float* conv2_b = (const float*)d_in[7];
  const float* ln1_w   = (const float*)d_in[8];
  const float* q_w     = (const float*)d_in[9];
  const float* k_w     = (const float*)d_in[10];
  const float* v_w     = (const float*)d_in[11];
  const float* o_w     = (const float*)d_in[12];
  const float* ln2_w   = (const float*)d_in[13];
  const float* fc1_w   = (const float*)d_in[14];
  const float* fc2_w   = (const float*)d_in[15];
  const float* fln_w   = (const float*)d_in[16];

  // -------- workspace carve-up (all sub-buffers 16B aligned) --------
  int* nf = (int*)d_ws;
  char* p = (char*)d_ws + 256;
  float* xbuf = (float*)p;  p += (size_t)B_ * TENC_ * HID_ * 4;          // residual (fp32)
  float* qbuf = (float*)p;  p += (size_t)B_ * TENC_ * HID_ * 4;
  float* kbuf = (float*)p;  p += (size_t)B_ * TENC_ * NKV_ * HD_ * 4;
  float* vbuf = (float*)p;  p += (size_t)B_ * TENC_ * NKV_ * HD_ * 4;
  __bf16* actA = (__bf16*)p; p += (size_t)B_ * TF_ * HID_ * 2;           // x1 / h+o temps
  __bf16* actB = (__bf16*)p; p += (size_t)B_ * T1_ * C1_ * 2;           // conv1 out / ffn hidden
  __bf16* w1b  = (__bf16*)p; p += (size_t)C1_ * HID_ * 5 * 2;
  __bf16* w2b  = (__bf16*)p; p += (size_t)HID_ * C1_ * 5 * 2;
  __bf16* qwb  = (__bf16*)p; p += (size_t)NL_ * HID_ * HID_ * 2;
  __bf16* kwb  = (__bf16*)p; p += (size_t)NL_ * NKV_ * HD_ * HID_ * 2;
  __bf16* vwb  = (__bf16*)p; p += (size_t)NL_ * NKV_ * HD_ * HID_ * 2;
  __bf16* owb  = (__bf16*)p; p += (size_t)NL_ * HID_ * HID_ * 2;
  __bf16* f1wb = (__bf16*)p; p += (size_t)NL_ * FFN_ * HID_ * 2;
  __bf16* f2wb = (__bf16*)p; p += (size_t)NL_ * HID_ * FFN_ * 2;

  __bf16* x1bf  = actA;                                  // [B*Tf, HID]
  __bf16* hbf   = actA;                                  // [B*Tenc, HID]
  __bf16* obf   = actA + (size_t)B_ * TENC_ * HID_;      // [B*Tenc, NH*HD]
  __bf16* y1bf  = actB;                                  // [B*T1, C1]
  __bf16* ffnbf = actB;                                  // [B*Tenc, FFN]

  // -------- one-time (per call) weight conversion to bf16 --------
  auto cvt = [&](const float* s, __bf16* d, int n) {
    f32_to_bf16_kernel<<<(n + 255) / 256, 256, 0, stream>>>(s, d, n);
  };
  cvt(conv1_w, w1b, C1_ * HID_ * 5);
  cvt(conv2_w, w2b, HID_ * C1_ * 5);
  cvt(q_w,   qwb,  NL_ * HID_ * HID_);
  cvt(k_w,   kwb,  NL_ * NKV_ * HD_ * HID_);
  cvt(v_w,   vwb,  NL_ * NKV_ * HD_ * HID_);
  cvt(o_w,   owb,  NL_ * HID_ * HID_);
  cvt(fc1_w, f1wb, NL_ * FFN_ * HID_);
  cvt(fc2_w, f2wb, NL_ * HID_ * FFN_);

  nf_kernel<<<B_, 256, 0, stream>>>(pmask, nf);
  frontend_kernel<<<B_ * TF_, 320, 0, stream>>>(audio, lin_w, log_k, nf, x1bf);

  // conv1: 320 -> 640, k=5, s=2, causal, fused bias+mask+silu, bf16 out
  {
    dim3 g((C1_ + 63) / 64, (B_ * T1_ + 127) / 128);
    gemm_conv_wmma<<<g, 256, 0, stream>>>(x1bf, w1b, conv1_b, nullptr, nullptr, y1bf, nf + 4,
                                          B_, TF_, T1_, HID_, C1_, 5, 2, 4, HID_ * 5, 1);
  }
  // conv2: 640 -> 320, k=5, s=2, causal, fused bias+mask, fp32 out (residual)
  {
    dim3 g((HID_ + 63) / 64, (B_ * TENC_ + 127) / 128);
    gemm_conv_wmma<<<g, 256, 0, stream>>>(y1bf, w2b, conv2_b, nullptr, xbuf, nullptr, nf + 8,
                                          B_, T1_, TENC_, C1_, HID_, 5, 2, 4, C1_ * 5, 0);
  }

  const int LW[NL_] = {16, 16, 16, 16, 16, 16};
  const int RW[NL_] = {4, 4, 0, 0, 4, 4};
  dim3 gq((HID_ + 63) / 64, (B_ * TENC_ + 127) / 128);
  dim3 gkv((NKV_ * HD_ + 63) / 64, (B_ * TENC_ + 127) / 128);
  dim3 gf1((FFN_ + 63) / 64, (B_ * TENC_ + 127) / 128);

  for (int L = 0; L < NL_; ++L) {
    ln_kernel_bf<<<B_ * TENC_, 320, 0, stream>>>(xbuf, ln1_w + (size_t)L * HID_, hbf, 1e-5f);
    gemm_conv_wmma<<<gq, 256, 0, stream>>>(hbf, qwb + (size_t)L * HID_ * HID_, nullptr, nullptr,
                                           qbuf, nullptr, nullptr,
                                           B_, TENC_, TENC_, HID_, HID_, 1, 1, 0, HID_, 0);
    gemm_conv_wmma<<<gkv, 256, 0, stream>>>(hbf, kwb + (size_t)L * NKV_ * HD_ * HID_, nullptr, nullptr,
                                            kbuf, nullptr, nullptr,
                                            B_, TENC_, TENC_, HID_, NKV_ * HD_, 1, 1, 0, HID_, 0);
    gemm_conv_wmma<<<gkv, 256, 0, stream>>>(hbf, vwb + (size_t)L * NKV_ * HD_ * HID_, nullptr, nullptr,
                                            vbuf, nullptr, nullptr,
                                            B_, TENC_, TENC_, HID_, NKV_ * HD_, 1, 1, 0, HID_, 0);
    attn_kernel<<<(B_ * TENC_ * NH_ + 255) / 256, 256, 0, stream>>>(qbuf, kbuf, vbuf, obf,
                                                                    nf + 8, LW[L], RW[L]);
    gemm_conv_wmma<<<gq, 256, 0, stream>>>(obf, owb + (size_t)L * HID_ * HID_, nullptr, xbuf,
                                           xbuf, nullptr, nullptr,
                                           B_, TENC_, TENC_, HID_, HID_, 1, 1, 0, HID_, 0);
    ln_kernel_bf<<<B_ * TENC_, 320, 0, stream>>>(xbuf, ln2_w + (size_t)L * HID_, hbf, 1e-5f);
    gemm_conv_wmma<<<gf1, 256, 0, stream>>>(hbf, f1wb + (size_t)L * FFN_ * HID_, nullptr, nullptr,
                                            nullptr, ffnbf, nullptr,
                                            B_, TENC_, TENC_, HID_, FFN_, 1, 1, 0, HID_, 1);
    gemm_conv_wmma<<<gq, 256, 0, stream>>>(ffnbf, f2wb + (size_t)L * HID_ * FFN_, nullptr, xbuf,
                                           xbuf, nullptr, nullptr,
                                           B_, TENC_, TENC_, FFN_, HID_, 1, 1, 0, FFN_, 0);
  }
  ln_kernel_f32<<<B_ * TENC_, 320, 0, stream>>>(xbuf, fln_w, (float*)d_out, 1e-5f);
}
